// Qwen3VL2BVisionLanguageAttentionKernel_59622736003269
// MI455X (gfx1250) — compile-verified, matrix operand order checked by reading, not run on hardware
//
#include <hip/hip_runtime.h>

// ---------------------------------------------------------------------------
// Problem constants (from reference): D=2048, H=16, HD=128, B=2, P=1024, S=2048
// ---------------------------------------------------------------------------
#define D_MODEL 2048
#define NHEADS  16
#define HDIM    128
#define BATCH   2
#define P_TOK   1024
#define S_TOK   2048
#define QK_SCALE 0.08838834764831845f  // 128^-0.5

typedef __attribute__((ext_vector_type(16))) __bf16       v16bf;
typedef __attribute__((ext_vector_type(8)))  float        v8f;
typedef __attribute__((ext_vector_type(4)))  unsigned int u32x4;

union Frag {
  v16bf   v;
  u32x4   q[2];
  __bf16  h[16];
};

__device__ __forceinline__ __bf16 f2bf(float f) {
  unsigned int u = __builtin_bit_cast(unsigned int, f);
  unsigned int r = (u + 0x7fffu + ((u >> 16) & 1u)) >> 16;   // round-to-nearest-even
  unsigned short s = (unsigned short)r;
  return __builtin_bit_cast(__bf16, s);
}

__device__ __forceinline__ v8f v8f_zero() {
  v8f z = {0.f, 0.f, 0.f, 0.f, 0.f, 0.f, 0.f, 0.f};
  return z;
}

// gfx1250 async copy: global memory -> LDS, 16 bytes per lane, ASYNCcnt-tracked.
// lds = wave-relative LDS byte offset (low 32 bits of the generic shared ptr),
// gptr = 64-bit global address (VGPR pair).
__device__ __forceinline__ void async_copy_b128(unsigned lds, const void* gptr) {
  unsigned long long ga = (unsigned long long)gptr;
  asm volatile("global_load_async_to_lds_b128 %0, %1, off"
               :: "v"(lds), "v"(ga) : "memory");
}

__device__ __forceinline__ void wait_async0() {
  asm volatile("s_wait_asynccnt 0x0" ::: "memory");
}

// ---------------------------------------------------------------------------
// Kernel 1: fp32 -> bf16 weight conversion (vectorized by 4)
// ---------------------------------------------------------------------------
__global__ __launch_bounds__(256) void cvt_bf16_kernel(const float* __restrict__ src,
                                                       __bf16* __restrict__ dst, int n4) {
  int i = blockIdx.x * blockDim.x + threadIdx.x;
  int stride = gridDim.x * blockDim.x;
  for (; i < n4; i += stride) {
    const float4 f = reinterpret_cast<const float4*>(src)[i];
    __bf16 o[4] = {f2bf(f.x), f2bf(f.y), f2bf(f.z), f2bf(f.w)};
    *reinterpret_cast<ulong*>(dst + 4 * (size_t)i) = *reinterpret_cast<ulong*>(o);
  }
}

// ---------------------------------------------------------------------------
// Kernel 2: pos-embed add + RMSNorm.  One 256-thread block per token row.
//   outF: x + pos (fp32 residual),  outB: rmsnorm(x+pos)*w as bf16
// ---------------------------------------------------------------------------
__global__ __launch_bounds__(256) void posnorm_kernel(const float* __restrict__ feat,
                                                      const float* __restrict__ pos,
                                                      const float* __restrict__ w,
                                                      float* __restrict__ outF,
                                                      __bf16* __restrict__ outB,
                                                      int T) {
  __shared__ float red[8];
  const int row = blockIdx.x;        // 0 .. B*T-1
  const int t   = row % T;           // pos-embed row
  const int tid = threadIdx.x;
  const int lane = tid & 31, wave = tid >> 5;

  const size_t rbase = (size_t)row * D_MODEL;
  const size_t pbase = (size_t)t * D_MODEL;

  float x[8];
  float ss = 0.f;
  #pragma unroll
  for (int i = 0; i < 8; ++i) {
    int idx = tid + i * 256;
    x[i] = feat[rbase + idx] + pos[pbase + idx];
    ss += x[i] * x[i];
  }
  #pragma unroll
  for (int o = 16; o > 0; o >>= 1) ss += __shfl_xor(ss, o, 32);
  if (lane == 0) red[wave] = ss;
  __syncthreads();
  float tot = 0.f;
  #pragma unroll
  for (int j = 0; j < 8; ++j) tot += red[j];
  const float r = rsqrtf(tot * (1.0f / D_MODEL) + 1e-6f);

  #pragma unroll
  for (int i = 0; i < 8; ++i) {
    int idx = tid + i * 256;
    outF[rbase + idx] = x[i];
    outB[rbase + idx] = f2bf(x[i] * r * w[idx]);
  }
}

// ---------------------------------------------------------------------------
// Kernel 3: bf16 WMMA GEMM:  C[M,N] = A[M,K] (row-major) * W[N,K]^T (row-major)
//   8 waves / block, 128x64 block tile, 32x32 per-wave tile, K-step 32.
//   Double-buffered tile staging with GLOBAL_LOAD_ASYNC_TO_LDS_B128 (ASYNCcnt):
//   issue tile k+1 async, run WMMA on tile k, s_wait_asynccnt 0, barrier, swap.
//   mode 0: f32 out (+optional residual add, +optional bf16 mirror)
//   mode 1: bf16 head-split out   [H, T, HD]   (Q/K), with scale
//   mode 2: bf16 head-split-trans [H, HD, T]   (V)
// ---------------------------------------------------------------------------
__global__ __launch_bounds__(256) void gemm_bf16_kernel(
    const __bf16* __restrict__ A, const __bf16* __restrict__ W,
    int M, int N, int K,
    float* __restrict__ outF, __bf16* __restrict__ outB,
    const float* __restrict__ residual,
    float scale, int mode, int headT, __bf16* __restrict__ outHead) {

  __shared__ __bf16 As[2][128 * 40];   // 128x32 tile, row stride 40 halves (pad)
  __shared__ __bf16 Bs[2][64 * 40];    // 64x32 tile

  const int tid  = threadIdx.x;
  const int lane = tid & 31, wave = tid >> 5;
  const int waveM = wave & 3;       // 4 waves along M (32 rows each)
  const int waveN = wave >> 2;      // 2 waves along N (32 cols each)
  const int lidx  = lane & 15;
  const int half  = (lane < 16) ? 0 : 8;     // K-chunk offset per WMMA layout
  const int mbase = (lane < 16) ? 0 : 8;     // C-row offset per WMMA layout

  const int bm0 = blockIdx.x * 128;
  const int bn0 = blockIdx.y * 64;

  v8f acc[2][2];
  #pragma unroll
  for (int i = 0; i < 2; ++i)
    #pragma unroll
    for (int j = 0; j < 2; ++j) acc[i][j] = v8f_zero();

  const int ar = tid >> 1;                 // A loader: 128 rows, 16 halves/thread
  const int ac = (tid & 1) * 16;
  const int br = tid >> 2;                 // B loader: 64 rows, 8 halves/thread
  const int bc = (tid & 3) * 8;

  const __bf16* gA = A + (size_t)(bm0 + ar) * K + ac;
  const __bf16* gB = W + (size_t)(bn0 + br) * K + bc;
  const unsigned ldsA0 = (unsigned)(unsigned long long)(uintptr_t)&As[0][ar * 40 + ac];
  const unsigned ldsA1 = (unsigned)(unsigned long long)(uintptr_t)&As[1][ar * 40 + ac];
  const unsigned ldsB0 = (unsigned)(unsigned long long)(uintptr_t)&Bs[0][br * 40 + bc];
  const unsigned ldsB1 = (unsigned)(unsigned long long)(uintptr_t)&Bs[1][br * 40 + bc];

  // async-stage one 32-deep K tile into LDS buffer `buf`
  auto stage = [&](int buf, int k0) {
    const unsigned la = buf ? ldsA1 : ldsA0;
    const unsigned lb = buf ? ldsB1 : ldsB0;
    async_copy_b128(la,      gA + k0);
    async_copy_b128(la + 16, gA + k0 + 8);   // +16 bytes / +8 halves
    async_copy_b128(lb,      gB + k0);
  };

  stage(0, 0);
  wait_async0();
  __syncthreads();

  int cur = 0;
  for (int k0 = 0; k0 < K; k0 += 32) {
    if (k0 + 32 < K) stage(cur ^ 1, k0 + 32);   // overlap next tile with compute

    Frag af[2], bfr[2];
    const __bf16* Ab = As[cur];
    const __bf16* Bb = Bs[cur];
    #pragma unroll
    for (int mt = 0; mt < 2; ++mt) {
      const __bf16* p = &Ab[(waveM * 32 + mt * 16 + lidx) * 40 + half];
      af[mt].q[0] = *(const u32x4*)p;
      af[mt].q[1] = *(const u32x4*)(p + 16);
    }
    #pragma unroll
    for (int nt = 0; nt < 2; ++nt) {
      const __bf16* p = &Bb[(waveN * 32 + nt * 16 + lidx) * 40 + half];
      bfr[nt].q[0] = *(const u32x4*)p;
      bfr[nt].q[1] = *(const u32x4*)(p + 16);
    }
    #pragma unroll
    for (int mt = 0; mt < 2; ++mt)
      #pragma unroll
      for (int nt = 0; nt < 2; ++nt)
        acc[mt][nt] = __builtin_amdgcn_wmma_f32_16x16x32_bf16(
            false, af[mt].v, false, bfr[nt].v, (short)0, acc[mt][nt], false, false);

    wait_async0();       // this wave's async writes for the next tile are done
    __syncthreads();     // everyone's are done; safe to swap buffers
    cur ^= 1;
  }

  // epilogue
  #pragma unroll
  for (int mt = 0; mt < 2; ++mt) {
    #pragma unroll
    for (int nt = 0; nt < 2; ++nt) {
      #pragma unroll
      for (int r = 0; r < 8; ++r) {
        const int m = bm0 + waveM * 32 + mt * 16 + mbase + r;
        const int n = bn0 + waveN * 32 + nt * 16 + lidx;
        float v = acc[mt][nt][r] * scale;
        if (mode == 0) {
          if (residual) v += residual[(size_t)m * N + n];
          outF[(size_t)m * N + n] = v;
          if (outB) outB[(size_t)m * N + n] = f2bf(v);
        } else if (mode == 1) {
          const int h = n >> 7, d = n & 127;
          outHead[((size_t)h * headT + m) * HDIM + d] = f2bf(v);
        } else {
          const int h = n >> 7, d = n & 127;
          outHead[((size_t)h * HDIM + d) * headT + m] = f2bf(v);
        }
      }
    }
  }
}

// ---------------------------------------------------------------------------
// Kernel 4: cross-attention for one (batch) side.
//   Block = 32 query rows of one head; full [32 x Tk] f32 score matrix lives in
//   LDS (<= 256 KB of the WGP's 320 KB).  Exact softmax in LDS, probabilities
//   optionally streamed to global (v2l_attn output), then A*V via bf16 WMMA
//   reading probs straight back from LDS and Vt (K-contiguous) from global.
//   Q: [H,Tq,128] bf16 (pre-scaled)   K: [H,Tk,128] bf16   Vt: [H,128,Tk] bf16
//   ctx: [Tq, 2048] bf16 (heads merged)
// ---------------------------------------------------------------------------
extern __shared__ float sc[];   // [32][Tk]

__global__ __launch_bounds__(256) void attn_kernel(
    const __bf16* __restrict__ Q, const __bf16* __restrict__ K,
    const __bf16* __restrict__ Vt, float* __restrict__ probs,
    __bf16* __restrict__ ctx, int Tq, int Tk) {

  const int h   = blockIdx.y;
  const int q0  = blockIdx.x * 32;
  const int tid = threadIdx.x;
  const int lane = tid & 31, wave = tid >> 5;
  const int qhalf  = wave & 1;        // which 16 query rows
  const int colgrp = wave >> 1;       // 0..3: column group
  const int lidx   = lane & 15;
  const int half   = (lane < 16) ? 0 : 8;
  const int mbase  = (lane < 16) ? 0 : 8;

  // ---- preload Q fragments for full HD=128 (4 K-chunks of 32) ----
  Frag qa[4];
  {
    const __bf16* qp = Q + ((size_t)h * Tq + (q0 + qhalf * 16 + lidx)) * HDIM;
    #pragma unroll
    for (int kc = 0; kc < 4; ++kc) {
      qa[kc].q[0] = *(const u32x4*)(qp + kc * 32 + half);
      qa[kc].q[1] = *(const u32x4*)(qp + kc * 32 + half + 16);
    }
  }

  // ---- phase 1: scores = Q K^T into LDS ----
  for (int s0 = colgrp * 16; s0 < Tk; s0 += 64) {
    Frag kb[4];
    const __bf16* kp = K + ((size_t)h * Tk + (s0 + lidx)) * HDIM;
    #pragma unroll
    for (int kc = 0; kc < 4; ++kc) {
      kb[kc].q[0] = *(const u32x4*)(kp + kc * 32 + half);
      kb[kc].q[1] = *(const u32x4*)(kp + kc * 32 + half + 16);
    }
    v8f c = v8f_zero();
    #pragma unroll
    for (int kc = 0; kc < 4; ++kc)
      c = __builtin_amdgcn_wmma_f32_16x16x32_bf16(false, qa[kc].v, false, kb[kc].v,
                                                  (short)0, c, false, false);
    #pragma unroll
    for (int r = 0; r < 8; ++r)
      sc[(qhalf * 16 + mbase + r) * Tk + s0 + lidx] = c[r];
  }
  __syncthreads();

  // ---- phase 2: exact softmax, 8 threads per row ----
  {
    const int row = tid >> 3;
    const int sub = tid & 7;
    float* srow = sc + (size_t)row * Tk;
    float mx = -1e30f;
    for (int s = sub; s < Tk; s += 8) mx = fmaxf(mx, srow[s]);
    #pragma unroll
    for (int o = 1; o < 8; o <<= 1) mx = fmaxf(mx, __shfl_xor(mx, o, 32));
    float sum = 0.f;
    for (int s = sub; s < Tk; s += 8) {
      float e = __expf(srow[s] - mx);
      srow[s] = e;
      sum += e;
    }
    #pragma unroll
    for (int o = 1; o < 8; o <<= 1) sum += __shfl_xor(sum, o, 32);
    const float inv = 1.0f / sum;
    if (probs) {
      float* prow = probs + ((size_t)h * Tq + q0 + row) * Tk;
      for (int s = sub; s < Tk; s += 8) {
        float p = srow[s] * inv;
        srow[s] = p;
        prow[s] = p;
      }
    } else {
      for (int s = sub; s < Tk; s += 8) srow[s] *= inv;
    }
  }
  __syncthreads();

  // ---- phase 3: ctx = P V  (A from LDS probs, B from Vt, K-contiguous) ----
  v8f acc[2];
  acc[0] = v8f_zero();
  acc[1] = v8f_zero();
  const int n0 = colgrp * 32;
  for (int s0 = 0; s0 < Tk; s0 += 32) {
    Frag pa;
    const float* pr = sc + (size_t)(qhalf * 16 + lidx) * Tk + s0 + half;
    #pragma unroll
    for (int i = 0; i < 8; ++i) pa.h[i] = f2bf(pr[i]);
    #pragma unroll
    for (int i = 0; i < 8; ++i) pa.h[8 + i] = f2bf(pr[16 + i]);
    #pragma unroll
    for (int nt = 0; nt < 2; ++nt) {
      Frag vb;
      const __bf16* vp = Vt + ((size_t)h * HDIM + (n0 + nt * 16 + lidx)) * Tk + s0 + half;
      vb.q[0] = *(const u32x4*)vp;
      vb.q[1] = *(const u32x4*)(vp + 16);
      acc[nt] = __builtin_amdgcn_wmma_f32_16x16x32_bf16(false, pa.v, false, vb.v,
                                                        (short)0, acc[nt], false, false);
    }
  }
  #pragma unroll
  for (int nt = 0; nt < 2; ++nt)
    #pragma unroll
    for (int r = 0; r < 8; ++r) {
      const int m = q0 + qhalf * 16 + mbase + r;
      const int d = n0 + nt * 16 + lidx;
      ctx[(size_t)m * D_MODEL + h * HDIM + d] = f2bf(acc[nt][r]);
    }
}

// ---------------------------------------------------------------------------
// Host orchestration
// ---------------------------------------------------------------------------
extern "C" void kernel_launch(void* const* d_in, const int* in_sizes, int n_in,
                              void* d_out, int out_size, void* d_ws, size_t ws_size,
                              hipStream_t stream) {
  (void)in_sizes; (void)n_in; (void)out_size;

  const float* vfeat = (const float*)d_in[0];
  const float* lfeat = (const float*)d_in[1];
  const float* vpos  = (const float*)d_in[2];
  const float* tpos  = (const float*)d_in[3];
  const float* vnw   = (const float*)d_in[4];
  const float* lnw   = (const float*)d_in[5];
  const float* wF[9];
  for (int i = 0; i < 9; ++i) wF[i] = (const float*)d_in[6 + i];
  // wF: 0=w_vq 1=w_vk 2=w_vv 3=w_lq 4=w_lk 5=w_lv 6=w_v2l 7=w_l2v 8=w_out

  // ---- output layout (flat, return order) ----
  float* out       = (float*)d_out;
  float* fused_out = out;                                   // [2,3072,2048]
  float* vis_out   = fused_out + (size_t)BATCH * 3072 * D_MODEL;
  float* lang_out  = vis_out   + (size_t)BATCH * P_TOK * D_MODEL;
  float* attn_out  = lang_out  + (size_t)BATCH * S_TOK * D_MODEL;  // [2,16,1024,2048]

  // ---- workspace carve-up ----
  char* ws = (char*)d_ws;
  size_t off = 0;
  auto carve = [&](size_t bytes) -> void* {
    void* p = ws + off;
    off += (bytes + 255) & ~(size_t)255;
    return p;
  };
  const size_t WELEM = (size_t)D_MODEL * D_MODEL;
  __bf16* wB[9];
  for (int i = 0; i < 9; ++i) wB[i] = (__bf16*)carve(WELEM * 2);
  float*  vf  = (float*) carve((size_t)BATCH * P_TOK * D_MODEL * 4);
  float*  lf  = (float*) carve((size_t)BATCH * S_TOK * D_MODEL * 4);
  __bf16* vn  = (__bf16*)carve((size_t)BATCH * P_TOK * D_MODEL * 2);
  __bf16* ln  = (__bf16*)carve((size_t)BATCH * S_TOK * D_MODEL * 2);
  __bf16* vq  = (__bf16*)carve((size_t)BATCH * P_TOK * D_MODEL * 2);
  __bf16* vk  = (__bf16*)carve((size_t)BATCH * P_TOK * D_MODEL * 2);
  __bf16* vvt = (__bf16*)carve((size_t)BATCH * P_TOK * D_MODEL * 2);
  __bf16* lq  = (__bf16*)carve((size_t)BATCH * S_TOK * D_MODEL * 2);
  __bf16* lk  = (__bf16*)carve((size_t)BATCH * S_TOK * D_MODEL * 2);
  __bf16* lvt = (__bf16*)carve((size_t)BATCH * S_TOK * D_MODEL * 2);
  __bf16* v2lctx = (__bf16*)carve((size_t)BATCH * P_TOK * D_MODEL * 2);
  __bf16* l2vctx = (__bf16*)carve((size_t)BATCH * S_TOK * D_MODEL * 2);
  __bf16* fusedB = (__bf16*)carve((size_t)BATCH * 3072 * D_MODEL * 2);
  (void)ws_size;

  // allow 256 KB dynamic LDS for the attention kernel
  (void)hipFuncSetAttribute((const void*)attn_kernel,
                            hipFuncAttributeMaxDynamicSharedMemorySize, 262144);

  // ---- 1. convert weights to bf16 ----
  for (int i = 0; i < 9; ++i)
    cvt_bf16_kernel<<<4096, 256, 0, stream>>>(wF[i], wB[i], (int)(WELEM / 4));

  // ---- 2. pos-embed + RMSNorm ----
  posnorm_kernel<<<BATCH * P_TOK, 256, 0, stream>>>(vfeat, vpos, vnw, vf, vn, P_TOK);
  posnorm_kernel<<<BATCH * S_TOK, 256, 0, stream>>>(lfeat, tpos, lnw, lf, ln, S_TOK);

  const size_t PD = (size_t)P_TOK * D_MODEL;
  const size_t SD = (size_t)S_TOK * D_MODEL;

  for (int b = 0; b < BATCH; ++b) {
    const __bf16* vn_b = vn + b * PD;
    const __bf16* ln_b = ln + b * SD;

    // ---- 3. QKV projections (head-split epilogues; Q pre-scaled) ----
    dim3 gv(P_TOK / 128, D_MODEL / 64), gl(S_TOK / 128, D_MODEL / 64);
    gemm_bf16_kernel<<<gv, 256, 0, stream>>>(vn_b, wB[0], P_TOK, D_MODEL, D_MODEL,
        nullptr, nullptr, nullptr, QK_SCALE, 1, P_TOK, vq + b * PD);
    gemm_bf16_kernel<<<gv, 256, 0, stream>>>(vn_b, wB[1], P_TOK, D_MODEL, D_MODEL,
        nullptr, nullptr, nullptr, 1.0f, 1, P_TOK, vk + b * PD);
    gemm_bf16_kernel<<<gv, 256, 0, stream>>>(vn_b, wB[2], P_TOK, D_MODEL, D_MODEL,
        nullptr, nullptr, nullptr, 1.0f, 2, P_TOK, vvt + b * PD);
    gemm_bf16_kernel<<<gl, 256, 0, stream>>>(ln_b, wB[3], S_TOK, D_MODEL, D_MODEL,
        nullptr, nullptr, nullptr, QK_SCALE, 1, S_TOK, lq + b * SD);
    gemm_bf16_kernel<<<gl, 256, 0, stream>>>(ln_b, wB[4], S_TOK, D_MODEL, D_MODEL,
        nullptr, nullptr, nullptr, 1.0f, 1, S_TOK, lk + b * SD);
    gemm_bf16_kernel<<<gl, 256, 0, stream>>>(ln_b, wB[5], S_TOK, D_MODEL, D_MODEL,
        nullptr, nullptr, nullptr, 1.0f, 2, S_TOK, lvt + b * SD);

    // ---- 4. cross attention ----
    // vision -> language: probs [H,P,S] materialized into d_out
    attn_kernel<<<dim3(P_TOK / 32, NHEADS), 256, 32 * S_TOK * 4, stream>>>(
        vq + b * PD, lk + b * SD, lvt + b * SD,
        attn_out + (size_t)b * NHEADS * P_TOK * S_TOK, v2lctx + b * PD, P_TOK, S_TOK);
    // language -> vision: probs discarded
    attn_kernel<<<dim3(S_TOK / 32, NHEADS), 256, 32 * P_TOK * 4, stream>>>(
        lq + b * SD, vk + b * PD, vvt + b * PD,
        nullptr, l2vctx + b * SD, S_TOK, P_TOK);

    // ---- 5. output projections + residual; mirror bf16 into fused buffer ----
    __bf16* fused_b = fusedB + (size_t)b * 3072 * D_MODEL;
    gemm_bf16_kernel<<<gv, 256, 0, stream>>>(v2lctx + b * PD, wB[6],
        P_TOK, D_MODEL, D_MODEL,
        vis_out + b * PD, fused_b, vf + b * PD, 1.0f, 0, 0, nullptr);
    gemm_bf16_kernel<<<gl, 256, 0, stream>>>(l2vctx + b * SD, wB[7],
        S_TOK, D_MODEL, D_MODEL,
        lang_out + b * SD, fused_b + PD, lf + b * SD, 1.0f, 0, 0, nullptr);

    // ---- 6. final fused projection ----
    gemm_bf16_kernel<<<dim3(3072 / 128, D_MODEL / 64), 256, 0, stream>>>(
        fused_b, wB[8], 3072, D_MODEL, D_MODEL,
        fused_out + (size_t)b * 3072 * D_MODEL, nullptr, nullptr, 1.0f, 0, 0, nullptr);
  }
}
